// LinearAttentionLayer_42219528519742
// MI455X (gfx1250) — matrix operand order repaired and checked by reading, not compiled, not run on hardware
//
#include <hip/hip_runtime.h>
#include <hip/hip_bf16.h>

// ---------------------------------------------------------------------------
// LinearAttentionLayer for MI455X (gfx1250, wave32, WMMA).
//   D_MODEL=512, N_HEADS=8, D_HEAD=64, B=2, T=1024, C_TABLES=2, NUM_BITS=8,
//   PHI_SCALE=0.1, MAX_EXP=4, EPS=1e-6
// Pipeline (fused):
//   1) per-weight ternary scale s (mean|W| reduction)
//   2) ternary-quantize weights -> f16 packed in WMMA-B fragment layout
//   3) x -> f16
//   4) Q/K projections with fused phi-lookup epilogue -> f16 (WMMA 2x2 blocked)
//      V projection with fused f16-convert epilogue    -> f16
//   5) chunked causal linear attention, all-WMMA (chunk C=64, state in regs),
//      writes f16 directly
//   6) output projection GEMM (+bias) -> f32 d_out
// ---------------------------------------------------------------------------

typedef _Float16 v8h  __attribute__((ext_vector_type(8)));
typedef _Float16 v16h __attribute__((ext_vector_type(16)));
typedef float    v8f  __attribute__((ext_vector_type(8)));

#define DM   512
#define BT   2048          // B*T
#define NW   (512*512)
#define NX   (2048*512)

// ---- WMMA fragment loaders (wave32; contiguous k = 16*kg + e in A and B, not ISA table order) ----------
// A 16x32 f16: lane = m + 16*kg (m=row), elements e -> K = e+16*kg
__device__ __forceinline__ v16h load_afrag(const _Float16* base, int ld,
                                           int row0, int k0, int lane) {
  const int m = lane & 15, kg = lane >> 4;
  const _Float16* p = base + (size_t)(row0 + m) * ld + k0 + 16 * kg;
  v8h lo = *(const v8h*)(p);
  v8h hi = *(const v8h*)(p + 8);
  return __builtin_shufflevector(lo, hi, 0,1,2,3,4,5,6,7,8,9,10,11,12,13,14,15);
}
// B 32x16 f16 from layout L[n][k] (k contiguous): lane = n' + 16*kg,
// elements e -> K = k0 + 16*kg + e  => one contiguous 32B run per lane.
__device__ __forceinline__ v16h load_bfrag(const _Float16* base, int ld,
                                           int col0, int k0, int lane) {
  const int n = lane & 15, kg = lane >> 4;
  return *(const v16h*)(base + (size_t)(col0 + n) * ld + k0 + 16 * kg);
}
__device__ __forceinline__ v8f wmma_f16(v16h a, v16h b, v8f c) {
  return __builtin_amdgcn_wmma_f32_16x16x32_f16(false, a, false, b,
                                                (short)0, c, false, false);
}

// ---- phi lookup on a scalar (PHI_SCALE=0.1, C_TABLES=2) -------------------
__device__ __forceinline__ float phi_scalar(float x, const float* __restrict__ t) {
  int i0 = (int)rintf(x * 10.0f) + 128;
  int i1 = (int)rintf(x * 5.0f) + 128;
  i0 = i0 < 0 ? 0 : (i0 > 255 ? 255 : i0);
  i1 = i1 < 0 ? 0 : (i1 > 255 ? 255 : i1);
  return t[i0] + t[256 + i1];
}

// ---- 1) ternary scale: s = exp2(clip(round(log2(mean|W|)), -4, 4)) --------
__global__ void __launch_bounds__(256)
wscale_kernel(const float* w0, const float* w1, const float* w2,
              const float* w3, float* sOut) {
  const float* W = (blockIdx.x == 0) ? w0 : (blockIdx.x == 1) ? w1
                 : (blockIdx.x == 2) ? w2 : w3;
  __shared__ float red[256];
  float acc = 0.f;
  for (int i = threadIdx.x; i < NW; i += 256) acc += fabsf(W[i]);
  red[threadIdx.x] = acc;
  __syncthreads();
  for (int s = 128; s > 0; s >>= 1) {
    if (threadIdx.x < s) red[threadIdx.x] += red[threadIdx.x + s];
    __syncthreads();
  }
  if (threadIdx.x == 0) {
    float alpha = red[0] / (float)NW + 1e-12f;
    float e = fminf(fmaxf(rintf(log2f(alpha)), -4.f), 4.f);
    sOut[blockIdx.x] = exp2f(e);
  }
}

// ---- 2) ternary quantize + pack into WMMA-B fragment layout ---------------
// Y[.,e] = sum_d X[.,d]*Wt[e,d]  => B[k=d][n=e].  Packed so the GEMM reads
// one lane-contiguous 32B vector per fragment:
//   P[(((nt*16 + kb)*32 + lane) << 4) + el],  nt=e/16, kb=d/32,
//   lane = (e&15) + 16*((d&31)>>4), el = d&15.
__global__ void __launch_bounds__(256)
pack_kernel(const float* W, const float* sArr, int sIdx, _Float16* P) {
  int id = blockIdx.x * 256 + threadIdx.x;      // [0, 512*512)
  float s = sArr[sIdx];
  int e = id >> 9, d = id & 511;
  float wq = fminf(fmaxf(rintf(W[id] / s), -1.f), 1.f) * s;
  int nt = e >> 4, np = e & 15;
  int kb = d >> 5, kin = d & 31, kg = kin >> 4, el = kin & 15;
  int lane = np + 16 * kg;
  P[(((size_t)(nt * 16 + kb) * 32 + lane) << 4) + el] = (_Float16)wq;
}

// ---- 3) f32 -> f16 elementwise (x only) -----------------------------------
__global__ void __launch_bounds__(256)
cvt16_kernel(const float* in, _Float16* out, int n) {
  int i = blockIdx.x * 256 + threadIdx.x;
  if (i < n) out[i] = (_Float16)in[i];
}

// ---- 4/6) GEMM: C[M,512] = A[M,512]_f16 @ Bpacked, 2x2 register blocking --
// 8 waves/block, one 32x32 macro-tile per wave (4 WMMA tiles), K-loop of
// 16 x {2 A-frag loads, 2 B-frag loads, 4 x wmma.f32.16x16x32.f16}.
// Epilogue: EPI=0 f32 + bias; EPI=1 phi-lookup -> f16; EPI=2 convert -> f16.
template <int EPI>
__global__ void __launch_bounds__(256)
gemm_wmma_kernel(const _Float16* __restrict__ A, const _Float16* __restrict__ Bp,
                 const float* __restrict__ bias, const float* __restrict__ table,
                 void* __restrict__ Cout, int M) {
  const int tid = threadIdx.x, lane = tid & 31, wave = tid >> 5;
  const int tile = blockIdx.x * 8 + wave;        // 32x32 macro-tiles
  const int m32 = tile >> 4, n32 = tile & 15;    // 512/32 = 16 col macro-tiles
  const int row0 = m32 * 32, col0 = n32 * 32;
  if (row0 >= M) return;                         // wave-uniform

  v8f acc00 = {}, acc01 = {}, acc10 = {}, acc11 = {};
  for (int kb = 0; kb < 16; ++kb) {
    v16h a0 = load_afrag(A, DM, row0,      kb * 32, lane);
    v16h a1 = load_afrag(A, DM, row0 + 16, kb * 32, lane);
    v16h b0 = *(const v16h*)(Bp + (((size_t)(n32 * 2 + 0) * 16 + kb) * 32 + lane) * 16);
    v16h b1 = *(const v16h*)(Bp + (((size_t)(n32 * 2 + 1) * 16 + kb) * 32 + lane) * 16);
    acc00 = wmma_f16(a0, b0, acc00);
    acc01 = wmma_f16(a0, b1, acc01);
    acc10 = wmma_f16(a1, b0, acc10);
    acc11 = wmma_f16(a1, b1, acc11);
  }

  const int np = lane & 15, half = lane >> 4;
  const v8f* accs[4] = { &acc00, &acc01, &acc10, &acc11 };
#pragma unroll
  for (int i = 0; i < 2; ++i)
#pragma unroll
    for (int j = 0; j < 2; ++j) {
      const v8f& acc = *accs[i * 2 + j];
      const int col = col0 + j * 16 + np;
#pragma unroll
      for (int r = 0; r < 8; ++r) {
        const int row = row0 + i * 16 + r + 8 * half;
        const float v = acc[r];
        if (EPI == 0) {
          ((float*)Cout)[(size_t)row * DM + col] = v + bias[col];
        } else if (EPI == 1) {
          ((_Float16*)Cout)[(size_t)row * DM + col] = (_Float16)phi_scalar(v, table);
        } else {
          ((_Float16*)Cout)[(size_t)row * DM + col] = (_Float16)v;
        }
      }
    }
}

// ---- 5) chunked causal linear attention (one block per (b,h)) -------------
// chunk C=64:  Sc = mask(phiQ @ phiK^T);  out = phiQ @ S_prev + Sc @ V;
//              Z  = rowsum(Sc) + phiQ . kvec_prev;  S += phiK^T @ V.
// State master (f32) lives in per-wave accumulator registers (2 tiles/wave),
// with an f16 LDS mirror (S^T[j][d]) used as the B operand next chunk.
// Output written directly as f16 for the final projection.
__global__ void __launch_bounds__(256)
attn_kernel(const _Float16* __restrict__ phiQ, const _Float16* __restrict__ phiK,
            const _Float16* __restrict__ Vh, _Float16* __restrict__ out16) {
  const int b = blockIdx.x >> 3, h = blockIdx.x & 7;
  const int tid = threadIdx.x, lane = tid & 31, wave = tid >> 5;
  const int np = lane & 15, half = lane >> 4;

  __shared__ __align__(64) _Float16 sQ [64 * 64];   // phiQ [t][d]
  __shared__ __align__(64) _Float16 sK [64 * 64];   // phiK [t][d]  (B of scores)
  __shared__ __align__(64) _Float16 sKt[64 * 64];   // phiK^T [d][t](B of S-update)
  __shared__ __align__(64) _Float16 sVt[64 * 64];   // V^T [j][t]   (B of Sc@V, A of S-update)
  __shared__ __align__(64) _Float16 sSc[64 * 64];   // masked scores [t][s]
  __shared__ __align__(64) _Float16 sSt[64 * 64];   // f16 state mirror S^T [j][d]
  __shared__ float sZ[64];
  __shared__ float sKv[64];                          // running sum of phiK

  for (int i = tid; i < 64 * 64; i += 256) sSt[i] = (_Float16)0.f;
  if (tid < 64) sKv[tid] = 0.f;

  // per-wave f32 state tiles: ids wave*2, wave*2+1 in a 4x4 grid (jt, dt)
  v8f accS0 = {}, accS1 = {};
  const int jt0 = (wave * 2) >> 2,     dt0 = (wave * 2) & 3;
  const int jt1 = (wave * 2 + 1) >> 2, dt1 = (wave * 2 + 1) & 3;
  __syncthreads();

  for (int tc = 0; tc < 16; ++tc) {
    const size_t gbase = ((size_t)b * 1024 + tc * 64) * DM + (size_t)h * 64;
    // --- load chunk (coalesced), plus transposed copies ---
#pragma unroll 4
    for (int k = 0; k < 16; ++k) {
      int idx = k * 256 + tid;
      int t = idx >> 6, d = idx & 63;
      size_t g = gbase + (size_t)t * DM + d;
      _Float16 q = phiQ[g], kk = phiK[g], vv = Vh[g];
      sQ [t * 64 + d] = q;
      sK [t * 64 + d] = kk;
      sKt[d * 64 + t] = kk;
      sVt[d * 64 + t] = vv;
    }
    __syncthreads();

    // --- scores = phiQ @ phiK^T, causal mask (s <= t), store f16 ---
#pragma unroll
    for (int ti = 0; ti < 2; ++ti) {
      int id = wave * 2 + ti, mt = id >> 2, nt = id & 3;
      v8f acc = {};
#pragma unroll
      for (int kb = 0; kb < 2; ++kb) {
        v16h a  = load_afrag(sQ, 64, mt * 16, kb * 32, lane);
        v16h bf = load_bfrag(sK, 64, nt * 16, kb * 32, lane);
        acc = wmma_f16(a, bf, acc);
      }
#pragma unroll
      for (int r = 0; r < 8; ++r) {
        int row = mt * 16 + r + 8 * half, col = nt * 16 + np;
        sSc[row * 64 + col] = (_Float16)((col <= row) ? acc[r] : 0.f);
      }
    }
    __syncthreads();

    // --- out = phiQ @ S_prev + Sc @ V (registers), Z = rowsum + phiQ.kvec ---
    v8f accO[2];
#pragma unroll
    for (int ti = 0; ti < 2; ++ti) {
      int id = wave * 2 + ti, mt = id >> 2, nt = id & 3;
      v8f acc = {};
#pragma unroll
      for (int kb = 0; kb < 2; ++kb) {            // inter-chunk: K = d
        v16h a  = load_afrag(sQ,  64, mt * 16, kb * 32, lane);
        v16h bf = load_bfrag(sSt, 64, nt * 16, kb * 32, lane);
        acc = wmma_f16(a, bf, acc);
      }
#pragma unroll
      for (int kb = 0; kb < 2; ++kb) {            // intra-chunk: K = s
        v16h a  = load_afrag(sSc, 64, mt * 16, kb * 32, lane);
        v16h bf = load_bfrag(sVt, 64, nt * 16, kb * 32, lane);
        acc = wmma_f16(a, bf, acc);
      }
      accO[ti] = acc;
    }
    if (tid < 64) {
      float rs = 0.f, dot = 0.f;
      for (int s = 0; s < 64; ++s) rs  += (float)sSc[tid * 64 + s];
      for (int d = 0; d < 64; ++d) dot += (float)sQ [tid * 64 + d] * sKv[d];
      sZ[tid] = fmaxf(rs + dot, 1e-6f);
    }
    __syncthreads();   // S_prev/kvec fully consumed; sZ visible

    // --- state update: S^T[j][d] += V^T @ phiK (A = sVt, B = sKt) ---
#pragma unroll
    for (int kb = 0; kb < 2; ++kb) {
      v16h a  = load_afrag(sVt, 64, jt0 * 16, kb * 32, lane);
      v16h bf = load_bfrag(sKt, 64, dt0 * 16, kb * 32, lane);
      accS0 = wmma_f16(a, bf, accS0);
    }
#pragma unroll
    for (int kb = 0; kb < 2; ++kb) {
      v16h a  = load_afrag(sVt, 64, jt1 * 16, kb * 32, lane);
      v16h bf = load_bfrag(sKt, 64, dt1 * 16, kb * 32, lane);
      accS1 = wmma_f16(a, bf, accS1);
    }
#pragma unroll
    for (int r = 0; r < 8; ++r) {
      sSt[(jt0 * 16 + r + 8 * half) * 64 + dt0 * 16 + np] = (_Float16)accS0[r];
      sSt[(jt1 * 16 + r + 8 * half) * 64 + dt1 * 16 + np] = (_Float16)accS1[r];
    }
    if (tid < 64) {                                // kvec += colsum(phiK)
      float c = sKv[tid];
      for (int t = 0; t < 64; ++t) c += (float)sK[t * 64 + tid];
      sKv[tid] = c;
    }

    // --- divide by Z, store f16 to global ---
#pragma unroll
    for (int ti = 0; ti < 2; ++ti) {
      int id = wave * 2 + ti, mt = id >> 2, nt = id & 3;
#pragma unroll
      for (int r = 0; r < 8; ++r) {
        int row = mt * 16 + r + 8 * half;
        out16[gbase + (size_t)row * DM + nt * 16 + np] =
            (_Float16)(accO[ti][r] / sZ[row]);
      }
    }
    __syncthreads();
  }
}

// ---------------------------------------------------------------------------
extern "C" void kernel_launch(void* const* d_in, const int* in_sizes, int n_in,
                              void* d_out, int out_size, void* d_ws, size_t ws_size,
                              hipStream_t stream) {
  const float* x   = (const float*)d_in[0];
  const float* Wq  = (const float*)d_in[1];
  const float* Wk  = (const float*)d_in[2];
  const float* Wv  = (const float*)d_in[3];
  const float* Wo  = (const float*)d_in[4];
  const float* bo  = (const float*)d_in[5];
  const float* tab = (const float*)d_in[6];
  float* out = (float*)d_out;
  char* ws = (char*)d_ws;

  // workspace carve (~12.3 MB)
  size_t off = 0;
  float* sArr = (float*)(ws + off);           off += 256;
  _Float16* packQ = (_Float16*)(ws + off);    off += (size_t)NW * 2;
  _Float16* packK = (_Float16*)(ws + off);    off += (size_t)NW * 2;
  _Float16* packV = (_Float16*)(ws + off);    off += (size_t)NW * 2;
  _Float16* packO = (_Float16*)(ws + off);    off += (size_t)NW * 2;
  _Float16* x16   = (_Float16*)(ws + off);    off += (size_t)NX * 2;
  _Float16* pq16  = (_Float16*)(ws + off);    off += (size_t)NX * 2;
  _Float16* pk16  = (_Float16*)(ws + off);    off += (size_t)NX * 2;
  _Float16* v16   = (_Float16*)(ws + off);    off += (size_t)NX * 2;
  _Float16* att16 = (_Float16*)(ws + off);    off += (size_t)NX * 2;

  // 1) ternary scales
  wscale_kernel<<<4, 256, 0, stream>>>(Wq, Wk, Wv, Wo, sArr);
  // 2) quantize + pack weights
  pack_kernel<<<NW / 256, 256, 0, stream>>>(Wq, sArr, 0, packQ);
  pack_kernel<<<NW / 256, 256, 0, stream>>>(Wk, sArr, 1, packK);
  pack_kernel<<<NW / 256, 256, 0, stream>>>(Wv, sArr, 2, packV);
  pack_kernel<<<NW / 256, 256, 0, stream>>>(Wo, sArr, 3, packO);
  // 3) x -> f16
  cvt16_kernel<<<NX / 256, 256, 0, stream>>>(x, x16, NX);
  // 4) projections: 32x32/wave, fused epilogues.
  //    wave-tiles = (2048/32)*(512/32) = 1024; 8 waves/block -> 128 blocks.
  gemm_wmma_kernel<1><<<128, 256, 0, stream>>>(x16, packQ, nullptr, tab, pq16, BT);
  gemm_wmma_kernel<1><<<128, 256, 0, stream>>>(x16, packK, nullptr, tab, pk16, BT);
  gemm_wmma_kernel<2><<<128, 256, 0, stream>>>(x16, packV, nullptr, nullptr, v16, BT);
  // 5) chunked causal linear attention (WMMA), one block per (b,h)
  attn_kernel<<<16, 256, 0, stream>>>(pq16, pk16, v16, att16);
  // 6) output projection (WMMA, +bias) -> f32
  gemm_wmma_kernel<0><<<128, 256, 0, stream>>>(att16, packO, bo, nullptr, out, BT);
}
